// Decoder_13331578487330
// MI455X (gfx1250) — matrix operand order repaired and checked by reading, not compiled, not run on hardware
//
#include <hip/hip_runtime.h>
#include <hip/hip_bf16.h>

// ---------------------------------------------------------------------------
// CDNA5 (gfx1250) WMMA types & fragment loaders
// ---------------------------------------------------------------------------
typedef __attribute__((ext_vector_type(16))) _Float16 v16h;
typedef __attribute__((ext_vector_type(8)))  float    v8f;
typedef __attribute__((ext_vector_type(4)))  int      v4i_;

union FragU { v16h h; uint4 u[2]; };

// Async global->LDS staging (gfx1250).
#if defined(__has_builtin)
#if __has_builtin(__builtin_amdgcn_global_load_async_to_lds_b128) && \
    __has_builtin(__builtin_amdgcn_s_wait_asynccnt)
#define USE_ASYNC_LDS 1
#endif
#endif
#ifndef USE_ASYNC_LDS
#define USE_ASYNC_LDS 0
#endif

__device__ __forceinline__ void copy16_g2s(const _Float16* __restrict__ g,
                                           _Float16* __restrict__ s) {
#if USE_ASYNC_LDS
  // (AS1 int4* src, AS3 int4* dst, Imm offset, Imm cpol)
  __builtin_amdgcn_global_load_async_to_lds_b128(
      (__attribute__((address_space(1))) v4i_*)g,
      (__attribute__((address_space(3))) v4i_*)s, 0, 0);
#else
  *(uint4*)s = *(const uint4*)g;
#endif
}

__device__ __forceinline__ void stage_fence() {
#if USE_ASYNC_LDS
  __builtin_amdgcn_s_wait_asynccnt(0);
#endif
  __syncthreads();
}

// ---- fast transcendentals on the CDNA5 TRANS pipe (co-executable with WMMA)
#define LOG2E_F 1.4426950408889634f
__device__ __forceinline__ float fast_sigmoid(float x) {
  // 1/(1+e^-x); x<<0 -> exp2(+inf)=inf -> rcp(inf)=0 ; x>>0 -> exp2(-inf)=0 -> 1
  return __builtin_amdgcn_rcpf(1.0f + __builtin_amdgcn_exp2f(-LOG2E_F * x));
}
#if defined(__has_builtin) && __has_builtin(__builtin_amdgcn_tanhf)
__device__ __forceinline__ float fast_tanh(float x) { return __builtin_amdgcn_tanhf(x); }
#else
__device__ __forceinline__ float fast_tanh(float x) {
  // 1 - 2/(e^{2x}+1): x>>0 -> e=inf -> 1 ; x<<0 -> e=0 -> -1  (Inf-safe)
  float e = __builtin_amdgcn_exp2f(2.0f * LOG2E_F * x);
  return 1.0f - 2.0f * __builtin_amdgcn_rcpf(e + 1.0f);
}
#endif

// A-matrix 16x32 f16 fragment (ISA 7.12.2, "16-bit A-Matrix 16x32"):
// lane M = lane&15 ; VGPR0..3 hold K pairs {0..7}+8*half, VGPR4..7 {16..23}+8*half
__device__ __forceinline__ v16h load_a_frag(const _Float16* __restrict__ X, int ldX,
                                            int mBase, int kc, int lane) {
  const int row = mBase + (lane & 15);
  const int hl  = lane >> 4;
  const _Float16* p = X + (size_t)row * ldX + kc + 8 * hl;
  FragU f;
  f.u[0] = *(const uint4*)(p);        // K = kc+8h    .. +7  -> VGPR0..3
  f.u[1] = *(const uint4*)(p + 16);   // K = kc+16+8h .. +7  -> VGPR4..7
  return f.h;
}

// B-matrix 32x16 f16 fragment: lane N = lane&15 ; VGPR0..3 hold K {0..7}+8*half, VGPR4..7 {16..23}+8*half
// Column n of B == row n of W for out = X @ W^T. Works for global or LDS rows.
__device__ __forceinline__ v16h load_b_frag(const _Float16* __restrict__ W, int ldW,
                                            int nBase, int kc, int lane) {
  const int row = nBase + (lane & 15);
  const int hl  = lane >> 4;
  const _Float16* p = W + (size_t)row * ldW + kc + 8 * hl;
  FragU f;
  f.u[0] = *(const uint4*)(p);
  f.u[1] = *(const uint4*)(p + 16);
  return f.h;
}

// ---------------------------------------------------------------------------
// Prep kernels
// ---------------------------------------------------------------------------
__global__ void k_f32_to_f16(const float* __restrict__ src, _Float16* __restrict__ dst, int n) {
  int i = blockIdx.x * blockDim.x + threadIdx.x;
  if (i < n) dst[i] = (_Float16)src[i];
}

__global__ void k_vadd(const float* __restrict__ a, const float* __restrict__ b,
                       float* __restrict__ c, int n) {
  int i = blockIdx.x * blockDim.x + threadIdx.x;
  if (i < n) c[i] = a[i] + b[i];
}

__global__ void k_zero_f32(float* __restrict__ p, int n) {
  int i = blockIdx.x * blockDim.x + threadIdx.x;
  if (i < n) p[i] = 0.0f;
}

// Gather the bs diagonal [64,64] blocks of the [8192,8192] input -> f16 [8192,64]
__global__ void k_extract_blocks(const float* __restrict__ in, _Float16* __restrict__ Xin) {
  int idx = blockIdx.x * blockDim.x + threadIdx.x;     // 8192*64
  if (idx >= 8192 * 64) return;
  int r = idx >> 6, k = idx & 63;
  Xin[idx] = (_Float16)in[(size_t)r * 8192 + (size_t)(r >> 6) * 64 + k];
}

// ---------------------------------------------------------------------------
// Generic GEMM (N fixed = 64): out[m][n] = act( X @ W^T + bias ), f16 out.
// Workgroup = 8 waves; all waves share the full 64xK B panel. When USE_LDS,
// the panel is async-staged in LDS once; inner-loop B reads are ds_load.
// A fragments are prefetched one K-chunk ahead; K is compile-time.
// ---------------------------------------------------------------------------
template <int K, int USE_LDS>
__global__ __launch_bounds__(256)
void k_gemm_xwt_f16(const _Float16* __restrict__ X, int ldX,
                    const _Float16* __restrict__ W,            // [64, K] row-major
                    const float* __restrict__ bias,
                    _Float16* __restrict__ Out, int ldOut,
                    int doRelu) {
  extern __shared__ _Float16 sW[];                             // [64, K] when USE_LDS
  const int tid  = threadIdx.x;
  const int wave = tid >> 5;
  const int lane = tid & 31;
  const int mBase = (blockIdx.x * 8 + wave) * 16;

  if constexpr (USE_LDS) {
    constexpr int kq = K >> 3;             // uint4 chunks per row
    constexpr int chunks = 64 * kq;
    for (int i = tid; i < chunks; i += 256) {
      int r = i / kq, c = i - r * kq;
      copy16_g2s(W + (size_t)r * K + c * 8, sW + (size_t)r * K + c * 8);
    }
    stage_fence();
  }
  const _Float16* Bp = USE_LDS ? sW : W;

  v8f acc[4] = {v8f{}, v8f{}, v8f{}, v8f{}};
  v16h a = load_a_frag(X, ldX, mBase, 0, lane);
#pragma unroll 2
  for (int kc = 0; kc < K; kc += 32) {
    v16h an = a;
    if (kc + 32 < K) an = load_a_frag(X, ldX, mBase, kc + 32, lane);  // prefetch
#pragma unroll
    for (int t = 0; t < 4; ++t) {
      v16h b = load_b_frag(Bp, K, t * 16, kc, lane);
      acc[t] = __builtin_amdgcn_wmma_f32_16x16x32_f16(false, a, false, b,
                                                      (short)0, acc[t], false, false);
    }
    a = an;
  }

  const int col0 = lane & 15, hl = lane >> 4;
#pragma unroll
  for (int t = 0; t < 4; ++t) {
    int n = t * 16 + col0;
    float bv = bias[n];
#pragma unroll
    for (int v = 0; v < 8; ++v) {
      float x = acc[t][v] + bv;
      if (doRelu) x = fmaxf(x, 0.0f);
      Out[(size_t)(mBase + v + 8 * hl) * ldOut + n] = (_Float16)x;
    }
  }
}

// ---------------------------------------------------------------------------
// Fused LSTM step (zero initial state => f-gate dead):
//   gates = X @ W_ih^T + b ; c = sig(i)*tanh(g) ; h = sig(o)*tanh(c)
// Workgroup = 8 waves sharing ONE hidden strip (16 units): the 48 needed W
// rows (i/g/o) are async-staged in LDS once; each wave owns a different
// 16-row M tile. Three WMMA accumulator chains; A prefetched a chunk ahead.
// W is [2048, K] row-major (PyTorch gate order i,f,g,o); f rows skipped.
// ---------------------------------------------------------------------------
template <int K>
__global__ __launch_bounds__(256)
void k_lstm_gates(const _Float16* __restrict__ X, int ldX,
                  const _Float16* __restrict__ W,     // [2048, K]
                  const float* __restrict__ bias,     // [2048] = b_ih + b_hh
                  _Float16* __restrict__ H) {         // [M, 512] f16
  extern __shared__ _Float16 sW[];                    // [48, K]: i rows, g rows, o rows
  const int tid  = threadIdx.x;
  const int wave = tid >> 5;
  const int lane = tid & 31;
  const int bm   = blockIdx.x >> 5;                   // M super-tile (128 rows)
  const int hs   = blockIdx.x & 31;                   // hidden strip (16 units)
  const int mBase = (bm * 8 + wave) * 16;
  const int nBase = hs * 16;

  {  // stage i/g/o weight rows for this strip into LDS
    constexpr int kq = K >> 3;
    constexpr int chunks = 48 * kq;
    for (int i = tid; i < chunks; i += 256) {
      int r = i / kq, c = i - r * kq;
      int gate = r >> 4, n = r & 15;
      int grow = (gate == 0 ? 0 : (gate == 1 ? 1024 : 1536)) + nBase + n;
      copy16_g2s(W + (size_t)grow * K + c * 8, sW + (size_t)r * K + c * 8);
    }
    stage_fence();
  }

  v8f ai = v8f{}, ag = v8f{}, ao = v8f{};
  v16h a = load_a_frag(X, ldX, mBase, 0, lane);
#pragma unroll 2
  for (int kc = 0; kc < K; kc += 32) {
    v16h an = a;
    if (kc + 32 < K) an = load_a_frag(X, ldX, mBase, kc + 32, lane);  // prefetch
    v16h bi = load_b_frag(sW, K, 0,  kc, lane);
    v16h bg = load_b_frag(sW, K, 16, kc, lane);
    v16h bo = load_b_frag(sW, K, 32, kc, lane);
    ai = __builtin_amdgcn_wmma_f32_16x16x32_f16(false, a, false, bi, (short)0, ai, false, false);
    ag = __builtin_amdgcn_wmma_f32_16x16x32_f16(false, a, false, bg, (short)0, ag, false, false);
    ao = __builtin_amdgcn_wmma_f32_16x16x32_f16(false, a, false, bo, (short)0, ao, false, false);
    a = an;
  }

  const int n  = nBase + (lane & 15);
  const int hl = lane >> 4;
  const float Bi = bias[n], Bg = bias[1024 + n], Bo = bias[1536 + n];
#pragma unroll
  for (int v = 0; v < 8; ++v) {
    float c = fast_sigmoid(ai[v] + Bi) * fast_tanh(ag[v] + Bg);
    float h = fast_sigmoid(ao[v] + Bo) * fast_tanh(c);
    H[(size_t)(mBase + v + 8 * hl) * 512 + n] = (_Float16)h;
  }
}

// ---------------------------------------------------------------------------
// Final 2-wide head: rel = hid @ W_m2^T + b_m2   (one thread per row)
// ---------------------------------------------------------------------------
__global__ void k_mlp_out(const _Float16* __restrict__ Hid,
                          const float* __restrict__ Wm2,   // [2,64] row-major
                          const float* __restrict__ bm2,
                          float* __restrict__ out) {       // [8192,2]
  int r = blockIdx.x * blockDim.x + threadIdx.x;
  if (r >= 8192) return;
  float s0 = bm2[0], s1 = bm2[1];
#pragma unroll 8
  for (int k = 0; k < 64; ++k) {
    float h = (float)Hid[(size_t)r * 64 + k];
    s0 = fmaf(h, Wm2[k], s0);
    s1 = fmaf(h, Wm2[64 + k], s1);
  }
  out[(size_t)r * 2 + 0] = s0;
  out[(size_t)r * 2 + 1] = s1;
}

// ---------------------------------------------------------------------------
// Host-side launcher
// ---------------------------------------------------------------------------
extern "C" void kernel_launch(void* const* d_in, const int* in_sizes, int n_in,
                              void* d_out, int out_size, void* d_ws, size_t ws_size,
                              hipStream_t stream) {
  (void)in_sizes; (void)n_in; (void)out_size; (void)ws_size;

  const float* input = (const float*)d_in[0];
  const float* W_sp  = (const float*)d_in[2];
  const float* b_sp  = (const float*)d_in[3];
  const float* W_ih0 = (const float*)d_in[4];
  const float* b_ih0 = (const float*)d_in[5];
  const float* b_hh0 = (const float*)d_in[6];
  const float* W_ih1 = (const float*)d_in[7];
  const float* b_ih1 = (const float*)d_in[8];
  const float* b_hh1 = (const float*)d_in[9];
  const float* W_h2p = (const float*)d_in[10];
  const float* b_h2p = (const float*)d_in[11];
  const float* W_m1  = (const float*)d_in[12];
  const float* b_m1  = (const float*)d_in[13];
  const float* W_m2  = (const float*)d_in[14];
  const float* b_m2  = (const float*)d_in[15];
  float* out = (float*)d_out;

  // Workspace carve-out (~22.5 MB)
  char* ws = (char*)d_ws;
  size_t off = 0;
  auto carve = [&](size_t bytes) -> char* {
    char* p = ws + off;
    off += (bytes + 255) & ~(size_t)255;
    return p;
  };
  _Float16* Wsp_h  = (_Float16*)carve(64   * 64  * 2);
  _Float16* Wih0_h = (_Float16*)carve(2048 * 64  * 2);
  _Float16* Wih1_h = (_Float16*)carve(2048 * 512 * 2);
  _Float16* Wh2p_h = (_Float16*)carve(64   * 512 * 2);
  _Float16* Wm1_h  = (_Float16*)carve(64   * 64  * 2);
  float*    b0     = (float*)   carve(2048 * 4);
  float*    b1     = (float*)   carve(2048 * 4);
  _Float16* Xin_h  = (_Float16*)carve(8192 * 64  * 2);
  _Float16* Xsp_h  = (_Float16*)carve(8192 * 64  * 2);
  _Float16* H1_h   = (_Float16*)carve(8192 * 512 * 2);
  _Float16* H2_h   = (_Float16*)carve(8192 * 512 * 2);
  _Float16* Emb_h  = (_Float16*)carve(8192 * 64  * 2);
  _Float16* Hid_h  = (_Float16*)carve(8192 * 64  * 2);

  const size_t ldsG64  = 64 * 64 * 2;    // GEMM B panel, K=64   (8 KB)
  const size_t ldsL64  = 48 * 64 * 2;    // LSTM i/g/o panel, K=64  (6 KB)
  const size_t ldsL512 = 48 * 512 * 2;   // LSTM i/g/o panel, K=512 (48 KB)

  // --- one-time prep ---
  k_f32_to_f16<<<(4096    + 255) / 256, 256, 0, stream>>>(W_sp,  Wsp_h,  4096);
  k_f32_to_f16<<<(131072  + 255) / 256, 256, 0, stream>>>(W_ih0, Wih0_h, 131072);
  k_f32_to_f16<<<(1048576 + 255) / 256, 256, 0, stream>>>(W_ih1, Wih1_h, 1048576);
  k_f32_to_f16<<<(32768   + 255) / 256, 256, 0, stream>>>(W_h2p, Wh2p_h, 32768);
  k_f32_to_f16<<<(4096    + 255) / 256, 256, 0, stream>>>(W_m1,  Wm1_h,  4096);
  k_vadd<<<8, 256, 0, stream>>>(b_ih0, b_hh0, b0, 2048);
  k_vadd<<<8, 256, 0, stream>>>(b_ih1, b_hh1, b1, 2048);
  k_zero_f32<<<256, 256, 0, stream>>>(out + 30 * 8192 * 2, 2 * 64 * 512);  // h_final = 0
  k_extract_blocks<<<2048, 256, 0, stream>>>(input, Xin_h);

  // spatial embedding: X0 = blocks @ W_sp^T + b_sp   [8192,64]
  k_gemm_xwt_f16<64, 1><<<64, 256, ldsG64, stream>>>(Xin_h, 64, Wsp_h, b_sp,
                                                     Xsp_h, 64, 0);

  // --- S = 30 sequential decode steps ---
  const _Float16* Xcur = Xsp_h;
  for (int s = 0; s < 30; ++s) {
    // layer0: [8192,64] -> [8192,512]  (64 M-supertiles x 32 strips = 2048 blocks)
    k_lstm_gates<64><<<2048, 256, ldsL64, stream>>>(Xcur, 64, Wih0_h, b0, H1_h);
    // layer1: [8192,512] -> [8192,512]  (dominant GEMM, LDS-staged weights)
    k_lstm_gates<512><<<2048, 256, ldsL512, stream>>>(H1_h, 512, Wih1_h, b1, H2_h);
    // hidden2pos: emb = H2 @ W_h2p^T + b  [8192,64]  (B from L2; 64KB panel > LDS cap)
    k_gemm_xwt_f16<512, 0><<<64, 256, 0, stream>>>(H2_h, 512, Wh2p_h, b_h2p,
                                                   Emb_h, 64, 0);
    // MLP layer1: hid = relu(emb @ W_m1^T + b)  [8192,64]
    k_gemm_xwt_f16<64, 1><<<64, 256, ldsG64, stream>>>(Emb_h, 64, Wm1_h, b_m1,
                                                       Hid_h, 64, 1);
    // head: traj[s] = hid @ W_m2^T + b  [8192,2]
    k_mlp_out<<<32, 256, 0, stream>>>(Hid_h, W_m2, b_m2, out + (size_t)s * 8192 * 2);

    Xcur = Emb_h;   // next step's input is this step's emb
  }
}
